// P1Level_39290360824050
// MI455X (gfx1250) — matrix operand order repaired and checked by reading, not compiled, not run on hardware
//
#include <hip/hip_runtime.h>

#define BT     32
#define Hdim   128
#define Wdim   128
#define Cdim   64
#define Npix   (Hdim * Wdim)     // 16384
#define Mseg   256
#define NSTEPS 40

typedef __attribute__((ext_vector_type(16))) _Float16 v16h;
typedef __attribute__((ext_vector_type(8)))  float    v8f;

// ---------------------------------------------------------------------------
// Kernel 1: per-pixel squared norm (exact f32)
// ---------------------------------------------------------------------------
__global__ void norm_kernel(const float* __restrict__ feat,
                            float* __restrict__ norms) {
    int g = blockIdx.x * blockDim.x + threadIdx.x;
    if (g >= BT * Npix) return;
    const float* f = feat + (size_t)g * Cdim;
    float s = 0.f;
    #pragma unroll
    for (int c = 0; c < Cdim; ++c) s += f[c] * f[c];
    norms[g] = s;
}

// ---------------------------------------------------------------------------
// Kernel 2: edge masks via WMMA.  One wave per 16-pixel row tile.
// dist^2(m, nb) = |f_m|^2 + |f_nb|^2 - 2 * dot(f_m, f_nb); dot via
// v_wmma_f32_16x16x32_f16 cross-Gram (diagonal), K=64 in two steps.
// edge bit j set iff neighbor in-bounds and dist^2 < 1.  j==4 (self): dist~0.
// ---------------------------------------------------------------------------
__global__ void edges_wmma_kernel(const float* __restrict__ feat,
                                  const float* __restrict__ norms,
                                  unsigned int* __restrict__ edges) {
    const int wave = (blockIdx.x * blockDim.x + threadIdx.x) >> 5;   // wave32
    const int lane = threadIdx.x & 31;
    const int tiles_per_row = Wdim / 16;                  // 8
    const int tiles_per_img = Hdim * tiles_per_row;       // 1024
    const int img = wave / tiles_per_img;
    if (img >= BT) return;                                // wave-uniform
    const int t  = wave % tiles_per_img;
    const int y  = t / tiles_per_row;
    const int x0 = (t % tiles_per_row) * 16;

    const float* fimg = feat  + (size_t)img * Npix * Cdim;
    const float* nimg = norms + (size_t)img * Npix;

    // ---- A fragments (center pixels), documented 16-bit A layout ----
    const int m     = lane & 15;
    const int kbase = (lane < 16) ? 0 : 8;
    const float* arow = fimg + ((size_t)y * Wdim + (x0 + m)) * Cdim;
    v16h a0, a1;
    #pragma unroll
    for (int e = 0; e < 16; ++e) {
        int k = kbase + (e < 8 ? e : e + 8);
        a0[e] = (_Float16)arow[k];
        a1[e] = (_Float16)arow[32 + k];
    }

    // classification owner lanes: diag (m,m) -> lane m (vgpr m) for m<8,
    // lane m+16 (vgpr m-8) for m>=8  => lanes {0..7, 24..31}
    const bool owner = (lane < 8) || (lane >= 24);
    const int  mo    = (lane < 8) ? lane : (lane - 16);   // owned pixel
    const int  ro    = (lane < 8) ? lane : (lane - 24);   // owned acc VGPR
    float nA = owner ? nimg[y * Wdim + x0 + mo] : 0.f;
    unsigned int mask = 0u;

    const int n     = lane & 15;                          // B column
    const int khalf = (lane < 16) ? 0 : 16;               // B K-half

    for (int j = 0; j < 9; ++j) {
        if (j == 4) { mask |= (owner ? (1u << 4) : 0u); continue; } // self edge
        const int dy = j / 3 - 1, dx = j % 3 - 1;

        const int by = y + dy, bx = x0 + n + dx;
        const bool inb = (by >= 0) && (by < Hdim) && (bx >= 0) && (bx < Wdim);
        const float* brow = fimg + ((size_t)(inb ? by : 0) * Wdim +
                                    (inb ? bx : 0)) * Cdim;
        v16h b0, b1;
        #pragma unroll
        for (int e = 0; e < 16; ++e) {
            int k = (khalf >> 1) + (e < 8 ? e : e + 8);
            b0[e] = (_Float16)(inb ? brow[k]      : 0.f);
            b1[e] = (_Float16)(inb ? brow[32 + k] : 0.f);
        }

        v8f acc = {};
        acc = __builtin_amdgcn_wmma_f32_16x16x32_f16(
                  false, a0, false, b0, (short)0, acc, false, false);
        acc = __builtin_amdgcn_wmma_f32_16x16x32_f16(
                  false, a1, false, b1, (short)0, acc, false, false);

        // extract owned diagonal element
        float diag = 0.f;
        #pragma unroll
        for (int r = 0; r < 8; ++r) diag = (r == ro) ? acc[r] : diag;

        if (owner && inb) {   // owner lane's (by,bx) is its own pixel's nbr
            float d2 = nA + nimg[by * Wdim + bx] - 2.f * diag;
            if (d2 < 1.0f) mask |= (1u << j);
        }
    }

    if (owner) edges[(size_t)img * Npix + y * Wdim + x0 + mo] = mask;
}

// ---------------------------------------------------------------------------
// Kernel 3: 40 rounds of synchronous min-label flooding, whole image in LDS.
// Dynamic LDS: int labA[N] + int labB[N] + ushort em[N]  = 160 KB.
// ---------------------------------------------------------------------------
extern "C" __global__ void __launch_bounds__(1024)
prop_kernel(const unsigned int* __restrict__ edges,
            int* __restrict__ labels) {
    extern __shared__ char smem[];
    int* labA = (int*)smem;
    int* labB = labA + Npix;
    unsigned short* em = (unsigned short*)(labB + Npix);

    const int img = blockIdx.x;
    const int tid = threadIdx.x, bs = blockDim.x;

    for (int p = tid; p < Npix; p += bs) {
        labA[p] = p;
        em[p]   = (unsigned short)edges[(size_t)img * Npix + p];
    }
    __syncthreads();

    int* src = labA; int* dst = labB;
    for (int it = 0; it < NSTEPS; ++it) {
        for (int p = tid; p < Npix; p += bs) {
            int best = src[p];
            unsigned msk = em[p];
            #pragma unroll
            for (int j = 0; j < 9; ++j) {
                if (j == 4) continue;
                if (msk & (1u << j)) {        // set only for in-bounds nbrs
                    int dy = j / 3 - 1, dx = j % 3 - 1;
                    int v = src[p + dy * Wdim + dx];
                    best = (v < best) ? v : best;
                }
            }
            dst[p] = best;
        }
        __syncthreads();
        int* tmp = src; src = dst; dst = tmp;
    }
    for (int p = tid; p < Npix; p += bs)
        labels[(size_t)img * Npix + p] = src[p];
}

// ---------------------------------------------------------------------------
// Kernel 4: relabel to contiguous ids capped at Mseg-1 (presence + cumsum).
// Dynamic LDS: int pres[N] + int part[1024] = 68 KB.
// ---------------------------------------------------------------------------
extern "C" __global__ void __launch_bounds__(1024)
relabel_kernel(const int* __restrict__ labels,
               int* __restrict__ seg) {
    extern __shared__ char smem[];
    int* pres = (int*)smem;
    int* part = pres + Npix;

    const int img = blockIdx.x;
    const int tid = threadIdx.x, bs = blockDim.x;   // bs == 1024
    const int* lab = labels + (size_t)img * Npix;

    for (int p = tid; p < Npix; p += bs) pres[p] = 0;
    __syncthreads();
    for (int p = tid; p < Npix; p += bs) pres[lab[p]] = 1;
    __syncthreads();

    // per-thread chunk sums (16 elems each)
    int s = 0;
    #pragma unroll
    for (int i = 0; i < 16; ++i) s += pres[tid * 16 + i];
    part[tid] = s;
    __syncthreads();

    // Hillis-Steele inclusive scan over 1024 partials
    for (int off = 1; off < 1024; off <<= 1) {
        int v = part[tid];
        if (tid >= off) v += part[tid - off];
        __syncthreads();
        part[tid] = v;
        __syncthreads();
    }

    // overwrite pres with inclusive cumsum (each thread owns its chunk)
    int run = (tid == 0) ? 0 : part[tid - 1];
    #pragma unroll
    for (int i = 0; i < 16; ++i) {
        int idx = tid * 16 + i;
        run += pres[idx];
        pres[idx] = run;
    }
    __syncthreads();

    for (int p = tid; p < Npix; p += bs) {
        int rank = pres[lab[p]] - 1;
        seg[(size_t)img * Npix + p] = (rank < Mseg - 1) ? rank : (Mseg - 1);
    }
}

// ---------------------------------------------------------------------------
// Kernel 5: per-segment mean/variance via LDS float atomics (ds_add_f32).
// Dynamic LDS: sums[256*64] + sqs[256*64] + cnt[256] = 129 KB.
// Output layout: [BT, 256, 128] = concat(mean(64), var(64)).
// ---------------------------------------------------------------------------
extern "C" __global__ void __launch_bounds__(1024)
agg_kernel(const float* __restrict__ feat,
           const int* __restrict__ seg,
           float* __restrict__ out) {
    extern __shared__ char smem[];
    float* sums = (float*)smem;
    float* sqs  = sums + Mseg * Cdim;
    float* cnt  = sqs  + Mseg * Cdim;

    const int img = blockIdx.x;
    const int tid = threadIdx.x, bs = blockDim.x;

    for (int i = tid; i < Mseg * Cdim; i += bs) { sums[i] = 0.f; sqs[i] = 0.f; }
    for (int i = tid; i < Mseg; i += bs) cnt[i] = 0.f;
    __syncthreads();

    for (int p = tid; p < Npix; p += bs) {
        int sgm = seg[(size_t)img * Npix + p];
        const float* f = feat + ((size_t)img * Npix + p) * Cdim;
        atomicAdd(&cnt[sgm], 1.0f);
        #pragma unroll
        for (int c = 0; c < Cdim; ++c) {
            float v = f[c];
            atomicAdd(&sums[sgm * Cdim + c], v);
            atomicAdd(&sqs[sgm * Cdim + c], v * v);
        }
    }
    __syncthreads();

    for (int i = tid; i < Mseg * Cdim; i += bs) {
        int sgm = i >> 6, c = i & 63;
        float d    = fmaxf(cnt[sgm], 1.0f);
        float mean = sums[i] / d;
        float var  = fmaxf(sqs[i] / d - mean * mean, 0.0f);
        float* o = out + ((size_t)img * Mseg + sgm) * (2 * Cdim);
        o[c]        = mean;
        o[Cdim + c] = var;
    }
}

// ---------------------------------------------------------------------------
extern "C" void kernel_launch(void* const* d_in, const int* in_sizes, int n_in,
                              void* d_out, int out_size, void* d_ws, size_t ws_size,
                              hipStream_t stream) {
    const float* feat = (const float*)d_in[0];
    float* out = (float*)d_out;

    char* ws = (char*)d_ws;
    const size_t npix_total = (size_t)BT * Npix;            // 524288
    float*        norms  = (float*)ws;                      // 2 MB
    unsigned int* edges  = (unsigned int*)(ws + npix_total * 4);   // 2 MB
    int*          labels = (int*)(ws + npix_total * 8);     // 2 MB
    int*          seg    = (int*)(ws + npix_total * 12);    // 2 MB

    // 1) norms
    {
        int nthreads = BT * Npix;
        norm_kernel<<<(nthreads + 255) / 256, 256, 0, stream>>>(feat, norms);
    }
    // 2) WMMA edge masks: one wave per 16-pixel tile
    {
        int waves = BT * Hdim * (Wdim / 16);                // 32768
        int threads = waves * 32;
        edges_wmma_kernel<<<threads / 256, 256, 0, stream>>>(feat, norms, edges);
    }
    // 3) label propagation, one workgroup per image, all 40 rounds in LDS
    {
        size_t lds = (size_t)Npix * 4 * 2 + (size_t)Npix * 2;   // 160 KB
        prop_kernel<<<BT, 1024, lds, stream>>>(edges, labels);
    }
    // 4) relabel (presence + cumsum in LDS)
    {
        size_t lds = (size_t)Npix * 4 + 1024 * 4;               // 68 KB
        relabel_kernel<<<BT, 1024, lds, stream>>>(labels, seg);
    }
    // 5) per-segment mean/var
    {
        size_t lds = (size_t)Mseg * Cdim * 4 * 2 + Mseg * 4;    // 129 KB
        agg_kernel<<<BT, 1024, lds, stream>>>(feat, seg, out);
    }
}
